// MahalanobisClassifier_5025111736976
// MI455X (gfx1250) — matrix operand order repaired and checked by reading, not compiled, not run on hardware
//
#include <hip/hip_runtime.h>

// ---------------- Types for WMMA ----------------
typedef __attribute__((ext_vector_type(16))) __bf16 v16bf;
typedef __attribute__((ext_vector_type(8)))  float  v8f;

// ---------------- Types for TDM descriptor groups ----------------
typedef __attribute__((ext_vector_type(4))) unsigned int u32x4;
typedef __attribute__((ext_vector_type(4))) int          i32x4;
typedef __attribute__((ext_vector_type(8))) int          i32x8;

union ABits { v16bf v; unsigned u[8]; };

__device__ __forceinline__ unsigned short f2bf(float f) {
    unsigned u = __builtin_bit_cast(unsigned, f);
    u += 0x7FFFu + ((u >> 16) & 1u);   // round-to-nearest-even
    return (unsigned short)(u >> 16);
}
__device__ __forceinline__ float bf2f(unsigned short h) {
    unsigned u = ((unsigned)h) << 16;
    return __builtin_bit_cast(float, u);
}

// Problem constants
#define TT 16384
#define CC 64
#define DD 256
#define ROWS 128           // rows of T per workgroup (8 waves x 16 rows)
#define PITCH 264          // LDS row pitch in bf16 elements (256 + 8 pad -> bank-conflict-free)
#define DIFF_BYTES (ROWS * PITCH * 2)          // 67,584
#define B_BYTES    (DD * PITCH * 2)            // 135,168
#define SMEM_BYTES (DIFF_BYTES + B_BYTES)      // 202,752  (<= 320KB WGP LDS)

#if defined(__has_builtin)
#if __has_builtin(__builtin_amdgcn_tensor_load_to_lds) && __has_builtin(__builtin_amdgcn_s_wait_tensorcnt)
#define USE_TDM 1
#endif
#endif

// ---------------- Kernel 1: P fp32 [c][d][e] -> bf16 transposed PT [c][e][d] ----------------
__global__ __launch_bounds__(256) void pconv_kernel(const float* __restrict__ P,
                                                    unsigned short* __restrict__ PT) {
    int i = blockIdx.x * 256 + threadIdx.x;       // 64*256*256 = 4,194,304 threads exactly
    int c = i >> 16;
    int rem = i & 0xFFFF;
    int d = rem >> 8;
    int e = rem & 255;
    PT[((size_t)c << 16) + (e << 8) + d] = f2bf(P[i]);
}

// ---------------- Kernel 2: main WMMA kernel ----------------
// grid = (T/ROWS, C), block = 256 threads (8 waves), dynamic LDS = SMEM_BYTES
__global__ __launch_bounds__(256) void mahal_wmma_kernel(const float* __restrict__ x,
                                                         const float* __restrict__ mu,
                                                         const unsigned short* __restrict__ PT,
                                                         float* __restrict__ out) {
    extern __shared__ __align__(16) unsigned char smem_raw[];
    unsigned short* diffS = (unsigned short*)smem_raw;                 // [ROWS][PITCH] bf16
    unsigned short* bS    = (unsigned short*)(smem_raw + DIFF_BYTES);  // [DD][PITCH] bf16  (bS[e][d])

    const int tid = threadIdx.x;
    const int cls = blockIdx.y;
    const int row_base = blockIdx.x * ROWS;
    const int w = tid >> 5;      // wave id 0..7

#ifdef USE_TDM
    // ---- Stage P_cls via Tensor Data Mover: one 2D tile 256x256 bf16, global->LDS,
    //      LDS pad 4 DWORDs after every 128 DWORDs (row pitch 256 -> 264 bf16). ----
    if (w == 0) {
        unsigned long long ga =
            (unsigned long long)(uintptr_t)(PT + ((size_t)cls << 16));
        unsigned lds_off = (unsigned)(uintptr_t)(void*)bS;

        u32x4 g0;
        g0.x = 1u;                                     // count=1, is_restore=0, gather off
        g0.y = lds_off;                                // lds_addr [63:32]
        g0.z = (unsigned)(ga & 0xFFFFFFFFull);         // global_addr [95:64]
        g0.w = (unsigned)((ga >> 32) & 0x1FFFFFFull)   // global_addr [120:96]
             | (2u << 30);                             // type=2 ("image") [127:126]

        i32x8 g1;
        g1[0] = (int)((1u << 16)        // data_size = 1 (2 bytes)
                    | (1u << 20)        // pad_enable
                    | (6u << 22)        // pad_interval: 128 DWORDs (512B row)
                    | (3u << 25));      // pad_amount: 4 DWORDs (16B)
        g1[1] = (int)(256u << 16);      // tensor_dim0[15:0]=256 in bits[63:48]
        g1[2] = (int)(256u << 16);      // tensor_dim1[15:0]=256 in bits[95:80]
        g1[3] = (int)(256u << 16);      // tile_dim0=256 in bits[127:112]
        g1[4] = (int)256;               // tile_dim1=256 in bits[143:128]; tile_dim2=0
        g1[5] = (int)256;               // tensor_dim0_stride[31:0]=256
        g1[6] = 0;                      // dim0_stride[47:32]=0, dim1_stride[15:0]=0
        g1[7] = 0;                      // tensor_dim1_stride hi = 0

        i32x4 gz = {0, 0, 0, 0};        // groups 2/3 unused for 2D tensor
#if __clang_major__ >= 23
        i32x8 gz8 = {0, 0, 0, 0, 0, 0, 0, 0};
        __builtin_amdgcn_tensor_load_to_lds(g0, g1, gz, gz, gz8, 0);
#else
        __builtin_amdgcn_tensor_load_to_lds(g0, g1, gz, gz, 0);
#endif
    }
#endif

    // ---- Stage diff = mu[cls] - x[row] as bf16 into LDS (coalesced float4 reads),
    //      overlapping with the TDM transfer of P_cls ----
    #pragma unroll 4
    for (int it = 0; it < 32; ++it) {
        int idx = tid + it * 256;           // 0..8191 quads
        int r = idx >> 6;                   // 0..127
        int d = (idx & 63) << 2;            // 0..252
        float4 xx = *(const float4*)(x + (size_t)(row_base + r) * DD + d);
        float4 mm = *(const float4*)(mu + cls * DD + d);
        unsigned short h0 = f2bf(mm.x - xx.x);
        unsigned short h1 = f2bf(mm.y - xx.y);
        unsigned short h2 = f2bf(mm.z - xx.z);
        unsigned short h3 = f2bf(mm.w - xx.w);
        uint2 pk;
        pk.x = (unsigned)h0 | ((unsigned)h1 << 16);
        pk.y = (unsigned)h2 | ((unsigned)h3 << 16);
        *(uint2*)(diffS + r * PITCH + d) = pk;
    }

#ifdef USE_TDM
    if (w == 0) {
        __builtin_amdgcn_s_wait_tensorcnt(0);   // TDM tile landed in LDS
    }
#else
    // ---- Fallback staging: PT[cls] (bf16) -> bS rows with pitch, 16B chunks ----
    {
        const uint4* src = (const uint4*)(PT + ((size_t)cls << 16));
        #pragma unroll 4
        for (int it = 0; it < 32; ++it) {
            int idx = tid + it * 256;       // 0..8191 chunks of 8 bf16
            int e = idx >> 5;               // 0..255
            int q = idx & 31;
            uint4 v = src[(e << 5) + q];
            *(uint4*)(bS + e * PITCH + (q << 3)) = v;
        }
    }
#endif
    __syncthreads();

    const int lane = tid & 31;
    const int h    = lane >> 4;  // lane half
    const int n    = lane & 15;  // M index for A, N index for B/C

    // ---- Preload A fragments (16x32 bf16 per §7.12.2) for all 8 K-chunks ----
    unsigned afrag[8][8];
    #pragma unroll
    for (int kb = 0; kb < 8; ++kb) {
        #pragma unroll
        for (int v = 0; v < 8; ++v) {
            int k0 = (v < 4) ? (2 * v + 8 * h) : (16 + 2 * (v - 4) + 8 * h);
            afrag[kb][v] = *(const unsigned*)(diffS + (w * 16 + n) * PITCH + kb * 32 + k0);
        }
    }

    float acc[8];
    #pragma unroll
    for (int i = 0; i < 8; ++i) acc[i] = 0.0f;

    // ---- 16 N-chunks x 8 K-chunks of V_WMMA_F32_16X16X32_BF16 ----
    for (int eb = 0; eb < 16; ++eb) {
        v8f cfrag = {};
        #pragma unroll
        for (int kb = 0; kb < 8; ++kb) {
            ABits a, b;
            #pragma unroll
            for (int v = 0; v < 8; ++v) a.u[v] = afrag[kb][v];
            // B (32x16 bf16): lane half h covers K=8h..8h+7 and 16+8h..16+8h+7, column N = n; bS[e][d] so K,K+1 contiguous
            #pragma unroll
            for (int v = 0; v < 8; ++v)
                b.u[v] = *(const unsigned*)(bS + (eb * 16 + n) * PITCH + kb * 32 + ((v < 4) ? (2 * v + 8 * h) : (16 + 2 * (v - 4) + 8 * h)));
            cfrag = __builtin_amdgcn_wmma_f32_16x16x32_bf16(
                false, a.v, false, b.v, (short)0, cfrag, false, false);
        }
        // Y tile elementwise * diff, accumulate: C/D layout lane(h,n) vgpr i -> (M=i+8h, N=n)
        #pragma unroll
        for (int i = 0; i < 8; ++i) {
            float dv = bf2f(diffS[(w * 16 + i + 8 * h) * PITCH + eb * 16 + n]);
            acc[i] += cfrag[i] * dv;
        }
    }

    // ---- Reduce over the 16 lanes of each half (N-dimension sum) ----
    #pragma unroll
    for (int i = 0; i < 8; ++i) {
        #pragma unroll
        for (int off = 1; off < 16; off <<= 1)
            acc[i] += __shfl_xor(acc[i], off, 32);
    }
    if (n == 0) {   // lane 0 writes rows 0..7, lane 16 writes rows 8..15
        #pragma unroll
        for (int i = 0; i < 8; ++i) {
            int t = row_base + w * 16 + i + 8 * h;
            out[(size_t)t * CC + cls] = -acc[i];
        }
    }
}

// ---------------- Fallback (workspace too small): exact fp32 VALU path ----------------
__global__ __launch_bounds__(256) void mahal_naive_kernel(const float* __restrict__ x,
                                                          const float* __restrict__ mu,
                                                          const float* __restrict__ P,
                                                          float* __restrict__ out) {
    int t = blockIdx.x;
    int c = blockIdx.y;
    int tid = threadIdx.x;
    __shared__ float diff[DD];
    __shared__ float red[256];
    diff[tid] = mu[c * DD + tid] - x[(size_t)t * DD + tid];
    __syncthreads();
    const float* Pr = P + ((size_t)c << 16) + (size_t)tid * DD;
    float s = 0.0f;
    #pragma unroll 4
    for (int e = 0; e < DD; ++e) s += Pr[e] * diff[e];
    s *= diff[tid];
    red[tid] = s;
    __syncthreads();
    for (int st = 128; st > 0; st >>= 1) {
        if (tid < st) red[tid] += red[tid + st];
        __syncthreads();
    }
    if (tid == 0) out[(size_t)t * CC + c] = -red[0];
}

// ---------------- Host launcher ----------------
extern "C" void kernel_launch(void* const* d_in, const int* in_sizes, int n_in,
                              void* d_out, int out_size, void* d_ws, size_t ws_size,
                              hipStream_t stream) {
    const float* x  = (const float*)d_in[0];   // [T, D] fp32
    const float* mu = (const float*)d_in[1];   // [C, D] fp32
    const float* P  = (const float*)d_in[2];   // [C, D, D] fp32
    float* out = (float*)d_out;                // [T, C] fp32

    const size_t need = (size_t)CC * DD * DD * sizeof(unsigned short); // 8,388,608 B
    if (ws_size >= need && d_ws != nullptr) {
        unsigned short* PT = (unsigned short*)d_ws;
        pconv_kernel<<<(CC * DD * DD) / 256, 256, 0, stream>>>(P, PT);
        (void)hipFuncSetAttribute((const void*)mahal_wmma_kernel,
                                  hipFuncAttributeMaxDynamicSharedMemorySize,
                                  SMEM_BYTES);
        mahal_wmma_kernel<<<dim3(TT / ROWS, CC), 256, SMEM_BYTES, stream>>>(x, mu, PT, out);
    } else {
        mahal_naive_kernel<<<dim3(TT, CC), 256, 0, stream>>>(x, mu, P, out);
    }
}
